// GeometricManifoldNetwork_56470230008016
// MI455X (gfx1250) — matrix operand order repaired and checked by reading, not compiled, not run on hardware
//
#include <hip/hip_runtime.h>
#include <math.h>

// ---------------- types ----------------
typedef float  v8f    __attribute__((ext_vector_type(8)));
typedef __bf16 v16bf  __attribute__((ext_vector_type(16)));
typedef __bf16 bf16x8 __attribute__((ext_vector_type(8)));
typedef __bf16 bf16x4 __attribute__((ext_vector_type(4)));

union FragBF { v16bf v; bf16x8 h[2]; };

__device__ __forceinline__ __bf16 f2bf(float f) {
    union { float f; unsigned u; } x; x.f = f;
    unsigned r = x.u + 0x7fffu + ((x.u >> 16) & 1u);   // round-to-nearest-even
    unsigned short hs = (unsigned short)(r >> 16);
    return __builtin_bit_cast(__bf16, hs);
}

__device__ __forceinline__ float wredsum(float v) {
    #pragma unroll
    for (int o = 16; o > 0; o >>= 1) v += __shfl_xor(v, o, 32);
    return v;
}

// ---------------- problem sizes ----------------
constexpr long BATCH = 32768;
constexpr long S_DIST  = 8388608;                   // manifold_distance
constexpr long S_CURV  = 8388609;                   // curvature_magnitude
constexpr long S_TANG  = 8388610;                   // tangent_norm
constexpr long OUT_MP  = 8388611;                   // mp [B,256]
constexpr long S_GCX   = 16777219;                  // geometric_complexity

// ---------------- workspace layout (bytes) ----------------
constexpr size_t OFF_H     = 0;                       // bf16 h [B,512] = 32MB, reused as g1
constexpr size_t OFF_MPB   = 32ull << 20;             // bf16 mp [B,256] = 16MB
constexpr size_t OFF_P     = 48ull << 20;             // f32 mp_raw then P [B,256] = 32MB
constexpr size_t OFF_GF    = 80ull << 20;             // bf16 gf [B,256] = 16MB
constexpr size_t OFF_W1B   = 96ull << 20;             // bf16 W1^T [512,1024] = 1MB
constexpr size_t OFF_W2B   = 97ull << 20;             // bf16 W2^T [256,512] = 256KB
constexpr size_t OFF_METB  = OFF_W2B  + (256ull << 10); // bf16 metric^T [256,256] = 128KB
constexpr size_t OFF_W3B   = OFF_METB + (128ull << 10); // bf16 W3^T [512,256] = 256KB
constexpr size_t OFF_W4B   = OFF_W3B  + (256ull << 10); // bf16 W4^T [256,512] = 256KB
constexpr size_t OFF_CPART = 98ull << 20;             // 4096*256*4 = 4MB
constexpr size_t OFF_CENTER= 102ull << 20;            // 256*4
constexpr size_t OFF_CM    = OFF_CENTER + 1024;       // 256*4
constexpr size_t OFF_SPART = OFF_CM + 1024;           // 4096*3*4 = 48KB
constexpr size_t OFF_CONNP = OFF_SPART + (64ull << 10); // 1024*4

// ---------------- bf16 WMMA GEMM: C = act(A @ W + bias), W pre-transposed [N,K] ----------------
// MODE 0: +bias, ReLU, store bf16 ; MODE 1: +bias, store f32 ; MODE 2: no bias, store f32
constexpr int BM = 128, BN = 128, BK = 32, SA = 40;   // SA: LDS row stride (80B, 16B multiple)

template <int MODE, bool AF32>
__global__ __launch_bounds__(256)
void gemm_wmma(const void* __restrict__ Ap, const __bf16* __restrict__ Wt,
               const float* __restrict__ bias, void* __restrict__ Cp,
               int N, int K) {
    __shared__ __bf16 As[2][BM * SA];                 // double-buffered
    __shared__ __bf16 Bs[2][BN * SA];
    const int tid  = threadIdx.x;
    const int lane = tid & 31, w = tid >> 5;
    const int wr   = w >> 1,   wc = w & 1;            // 4x2 wave grid
    const int half = lane >> 4, l16 = lane & 15;
    const long rowBase = (long)blockIdx.y * BM;
    const int  nBase   = blockIdx.x * BN;

    v8f acc[2][4] = {};

    float4 aRegF[4];                                  // register-staged tiles
    bf16x8 aRegH[2];
    bf16x8 bReg[2];

    const int ar0 = tid >> 3, akq = (tid & 7) * 4;    // AF32 staging map (128 rows x 8 quads)
    const int hr0 = tid >> 2, hko = (tid & 3) * 8;    // bf16 staging map (128 rows x 4 octs)

    auto loadTiles = [&](int k0, bool pf) {
        if constexpr (AF32) {
            const float* A = (const float*)Ap;
            #pragma unroll
            for (int i = 0; i < 4; ++i) {
                const float* p = A + (rowBase + ar0 + i * 32) * K + k0 + akq;
                aRegF[i] = *(const float4*)p;
                if (pf) __builtin_prefetch(p + BK, 0, 0);      // global_prefetch_b8
            }
        } else {
            const __bf16* A = (const __bf16*)Ap;
            #pragma unroll
            for (int i = 0; i < 2; ++i) {
                const __bf16* p = A + (rowBase + hr0 + i * 64) * K + k0 + hko;
                aRegH[i] = *(const bf16x8*)p;
                if (pf) __builtin_prefetch(p + BK, 0, 0);
            }
        }
        #pragma unroll
        for (int i = 0; i < 2; ++i)                   // W^T: contiguous 16B loads
            bReg[i] = *(const bf16x8*)(Wt + (long)(nBase + hr0 + i * 64) * K + k0 + hko);
    };
    auto storeTiles = [&](int buf) {
        if constexpr (AF32) {
            #pragma unroll
            for (int i = 0; i < 4; ++i) {
                bf16x4 hv;
                hv[0] = f2bf(aRegF[i].x); hv[1] = f2bf(aRegF[i].y);
                hv[2] = f2bf(aRegF[i].z); hv[3] = f2bf(aRegF[i].w);
                *(bf16x4*)(&As[buf][(ar0 + i * 32) * SA + akq]) = hv;
            }
        } else {
            #pragma unroll
            for (int i = 0; i < 2; ++i)
                *(bf16x8*)(&As[buf][(hr0 + i * 64) * SA + hko]) = aRegH[i];
        }
        #pragma unroll
        for (int i = 0; i < 2; ++i)
            *(bf16x8*)(&Bs[buf][(hr0 + i * 64) * SA + hko]) = bReg[i];
    };
    auto compute = [&](int buf) {
        FragBF fa[2], fb[4];
        #pragma unroll
        for (int t = 0; t < 2; ++t) {
            const __bf16* ap = &As[buf][(wr * 32 + t * 16 + l16) * SA];
            fa[t].h[0] = *(const bf16x8*)(ap + half * 8);        // K 0..7  / 8..15
            fa[t].h[1] = *(const bf16x8*)(ap + 16 + half * 8);   // K 16..23 / 24..31
        }
        #pragma unroll
        for (int u = 0; u < 4; ++u) {
            const __bf16* bp = &Bs[buf][(wc * 64 + u * 16 + l16) * SA];
            fb[u].h[0] = *(const bf16x8*)(bp + half * 8);        // K 0..7  / 8..15
            fb[u].h[1] = *(const bf16x8*)(bp + 16 + half * 8);   // K 16..23 / 24..31
        }
        #pragma unroll
        for (int t = 0; t < 2; ++t)
            #pragma unroll
            for (int u = 0; u < 4; ++u)
                acc[t][u] = __builtin_amdgcn_wmma_f32_16x16x32_bf16(
                    false, fa[t].v, false, fb[u].v, (short)0, acc[t][u], false, false);
    };

    // pipelined main loop: one barrier per k-step, global latency hidden behind WMMA
    loadTiles(0, true);
    storeTiles(0);
    __syncthreads();
    int cur = 0;
    for (int k0 = BK; k0 < K; k0 += BK) {
        loadTiles(k0, k0 + BK < K);   // global loads in flight
        compute(cur);                  // WMMA on current buffer
        storeTiles(cur ^ 1);           // waits loads, fills other buffer
        __syncthreads();
        cur ^= 1;
    }
    compute(cur);

    // ---- epilogue ----
    #pragma unroll
    for (int u = 0; u < 4; ++u) {
        int nb = nBase + wc * 64 + u * 16 + l16;
        float bv = (MODE == 2) ? 0.0f : bias[nb];
        #pragma unroll
        for (int t = 0; t < 2; ++t) {
            long mrow = rowBase + wr * 32 + t * 16 + half * 8;
            #pragma unroll
            for (int r = 0; r < 8; ++r) {
                float v = acc[t][u][r] + bv;
                if (MODE == 0) v = fmaxf(v, 0.0f);
                long idx = (mrow + r) * N + nb;
                if (MODE == 0) ((__bf16*)Cp)[idx] = f2bf(v);
                else           ((float*)Cp)[idx]  = v;
            }
        }
    }
}

// ---------------- weight fp32 [K,N] -> bf16 transposed [N,K] ----------------
__global__ void cvt_w_t(const float* __restrict__ src, __bf16* __restrict__ dst, int K, int N) {
    long i = (long)blockIdx.x * 256 + threadIdx.x;
    if (i < (long)K * N) {
        int k = (int)(i / N), n = (int)(i % N);       // coalesced read
        dst[(long)n * K + k] = f2bf(src[i]);
    }
}

// ---------------- row L2-normalize + deterministic column partial sums ----------------
__global__ __launch_bounds__(256)
void rownorm_center(const float* __restrict__ mpraw, float* __restrict__ mp_out,
                    __bf16* __restrict__ mpb, float* __restrict__ cpart) {
    __shared__ float cs[8][256];
    int tid = threadIdx.x, lane = tid & 31, w = tid >> 5;
    long row = (long)blockIdx.x * 8 + w;
    const float* src = mpraw + row * 256 + lane * 8;
    float4 a = *(const float4*)src;
    float4 b = *(const float4*)(src + 4);
    float v[8] = {a.x, a.y, a.z, a.w, b.x, b.y, b.z, b.w};
    float ss = 0.f;
    #pragma unroll
    for (int i = 0; i < 8; ++i) ss += v[i] * v[i];
    ss = wredsum(ss);
    float scale = 1.0f / fmaxf(sqrtf(ss), 1e-12f);
    float* op = mp_out + row * 256 + lane * 8;        // 4B-aligned region in d_out
    bf16x8 hb;
    #pragma unroll
    for (int i = 0; i < 8; ++i) {
        v[i] *= scale;
        op[i] = v[i];
        hb[i] = f2bf(v[i]);
        cs[w][lane * 8 + i] = v[i];
    }
    *(bf16x8*)(mpb + row * 256 + lane * 8) = hb;
    __syncthreads();
    float s = 0.f;                                    // fixed-order: deterministic
    #pragma unroll
    for (int r = 0; r < 8; ++r) s += cs[r][tid];
    cpart[(long)blockIdx.x * 256 + tid] = s;
}

// ---------------- center = mean(mp,axis=0); cm = center @ metric ----------------
__global__ void reduce_center(const float* __restrict__ cpart, const float* __restrict__ metric,
                              float* __restrict__ center, float* __restrict__ cm, int nblocks) {
    __shared__ float cl[256];
    int c = threadIdx.x;
    float s = 0.f;
    for (int i = 0; i < nblocks; ++i) s += cpart[(long)i * 256 + c];
    s *= (1.0f / 32768.0f);
    cl[c] = s; center[c] = s;
    __syncthreads();
    float m = 0.f;
    for (int j = 0; j < 256; ++j) m += cl[j] * metric[j * 256 + c];
    cm[c] = m;
}

// ---------------- sum(|conn|) stage 1 ----------------
__global__ __launch_bounds__(256)
void conn_abs_partial(const float* __restrict__ conn, float* __restrict__ part) {
    __shared__ float red[256];
    long gtid = (long)blockIdx.x * 256 + threadIdx.x;
    long stride = (long)gridDim.x * 256;
    float s = 0.f;
    for (long i = gtid; i < 4194304; i += stride) {   // 256^3 / 4 float4's
        float4 f = ((const float4*)conn)[i];
        s += fabsf(f.x) + fabsf(f.y) + fabsf(f.z) + fabsf(f.w);
    }
    red[threadIdx.x] = s; __syncthreads();
    for (int o = 128; o > 0; o >>= 1) {
        if (threadIdx.x < o) red[threadIdx.x] += red[threadIdx.x + o];
        __syncthreads();
    }
    if (threadIdx.x == 0) part[blockIdx.x] = red[0];
}

// ---------------- per-row geometry: dist, tangent, gf; stat partials ----------------
__global__ __launch_bounds__(256)
void geom_rows(const float* __restrict__ mp, const float* __restrict__ P,
               const float* __restrict__ center, const float* __restrict__ cm,
               __bf16* __restrict__ gfb, float* __restrict__ spart) {
    __shared__ float cen[256], cmv[256];
    __shared__ float st[8][3];
    int tid = threadIdx.x, lane = tid & 31, w = tid >> 5;
    cen[tid] = center[tid]; cmv[tid] = cm[tid];
    __syncthreads();
    long row = (long)blockIdx.x * 8 + w;
    const float* mpr = mp + row * 256 + lane * 8;
    const float* Pr  = P  + row * 256 + lane * 8;
    float mv[8], d[8], ce[8];
    float dd = 0.f, dc = 0.f;
    #pragma unroll
    for (int i = 0; i < 8; ++i) {
        mv[i] = mpr[i];
        ce[i] = cen[lane * 8 + i];
        d[i]  = mv[i] - ce[i];
        float q = Pr[i] - cmv[lane * 8 + i];          // diff@metric = mp@metric - center@metric
        dd += d[i] * q;
        dc += d[i] * ce[i];
    }
    dd = wredsum(dd); dc = wredsum(dc);
    float dist = sqrtf(fmaxf(dd, 0.0f));
    float tn2 = 0.f; bf16x8 hb;
    #pragma unroll
    for (int i = 0; i < 8; ++i) {
        float t = d[i] - dc * ce[i];                  // tangent
        tn2 += t * t;
        hb[i] = f2bf(mv[i] + 0.1f * t);               // gf
    }
    tn2 = wredsum(tn2);
    *(bf16x8*)(gfb + row * 256 + lane * 8) = hb;
    if (lane == 0) { st[w][0] = dist; st[w][1] = dist * dist; st[w][2] = sqrtf(tn2); }
    __syncthreads();
    if (tid < 3) {
        float s = 0.f;
        #pragma unroll
        for (int r = 0; r < 8; ++r) s += st[r][tid];  // fixed-order: deterministic
        spart[(long)blockIdx.x * 3 + tid] = s;
    }
}

// ---------------- scalars ----------------
__global__ void finalize_stats(const float* __restrict__ spart, const float* __restrict__ connp,
                               float* __restrict__ out, int nblk, int nconn) {
    if (threadIdx.x == 0 && blockIdx.x == 0) {
        float ds = 0.f, d2 = 0.f, tn = 0.f;
        for (int i = 0; i < nblk; ++i) {
            ds += spart[i * 3 + 0];
            d2 += spart[i * 3 + 1];
            tn += spart[i * 3 + 2];
        }
        float cs = 0.f;
        for (int i = 0; i < nconn; ++i) cs += connp[i];
        const float Bf = 32768.0f;
        out[S_DIST] = ds / Bf;
        out[S_CURV] = cs / 256.0f;                    // mean_m sum_{j,k}|conn[m,j,k]|
        out[S_TANG] = tn / Bf;
        float var = (d2 - ds * ds / Bf) / (Bf - 1.0f); // ddof=1
        out[S_GCX] = sqrtf(fmaxf(var, 0.0f));
    }
}

// ---------------- launch ----------------
extern "C" void kernel_launch(void* const* d_in, const int* in_sizes, int n_in,
                              void* d_out, int out_size, void* d_ws, size_t ws_size,
                              hipStream_t stream) {
    (void)in_sizes; (void)n_in; (void)out_size; (void)ws_size;
    const float* x      = (const float*)d_in[0];
    const float* W1     = (const float*)d_in[1];
    const float* b1     = (const float*)d_in[2];
    const float* W2     = (const float*)d_in[3];
    const float* b2     = (const float*)d_in[4];
    const float* metric = (const float*)d_in[5];
    const float* conn   = (const float*)d_in[6];
    const float* W3     = (const float*)d_in[7];
    const float* b3     = (const float*)d_in[8];
    const float* W4     = (const float*)d_in[9];
    const float* b4     = (const float*)d_in[10];

    char*  ws  = (char*)d_ws;
    float* out = (float*)d_out;

    __bf16* h    = (__bf16*)(ws + OFF_H);     // [B,512] bf16, later reused as g1
    __bf16* mpb  = (__bf16*)(ws + OFF_MPB);   // [B,256] bf16
    float*  Pf   = (float*)(ws + OFF_P);      // [B,256] f32: mp_raw, then P
    __bf16* gfb  = (__bf16*)(ws + OFF_GF);    // [B,256] bf16
    __bf16* w1t  = (__bf16*)(ws + OFF_W1B);
    __bf16* w2t  = (__bf16*)(ws + OFF_W2B);
    __bf16* mett = (__bf16*)(ws + OFF_METB);
    __bf16* w3t  = (__bf16*)(ws + OFF_W3B);
    __bf16* w4t  = (__bf16*)(ws + OFF_W4B);
    float*  cpart  = (float*)(ws + OFF_CPART);
    float*  center = (float*)(ws + OFF_CENTER);
    float*  cm     = (float*)(ws + OFF_CM);
    float*  spart  = (float*)(ws + OFF_SPART);
    float*  connp  = (float*)(ws + OFF_CONNP);

    dim3 blk(256);

    // weights -> bf16, transposed to [N,K] so GEMM B-staging is contiguous 16B loads
    cvt_w_t<<<(524288 + 255) / 256, blk, 0, stream>>>(W1, w1t, 1024, 512);
    cvt_w_t<<<(131072 + 255) / 256, blk, 0, stream>>>(W2, w2t, 512, 256);
    cvt_w_t<<<(65536  + 255) / 256, blk, 0, stream>>>(metric, mett, 256, 256);
    cvt_w_t<<<(131072 + 255) / 256, blk, 0, stream>>>(W3, w3t, 256, 512);
    cvt_w_t<<<(131072 + 255) / 256, blk, 0, stream>>>(W4, w4t, 512, 256);

    // encoder
    gemm_wmma<0, true ><<<dim3(512 / BN, BATCH / BM), blk, 0, stream>>>(x,   w1t, b1, h,   512, 1024);
    gemm_wmma<1, false><<<dim3(256 / BN, BATCH / BM), blk, 0, stream>>>(h,   w2t, b2, Pf,  256, 512);

    // normalize + batch mean (deterministic two-stage)
    rownorm_center<<<BATCH / 8, blk, 0, stream>>>(Pf, out + OUT_MP, mpb, cpart);
    reduce_center<<<1, blk, 0, stream>>>(cpart, metric, center, cm, BATCH / 8);

    // curvature reduction over conn (bandwidth-bound)
    conn_abs_partial<<<1024, blk, 0, stream>>>(conn, connp);

    // P = mp @ metric (overwrites mp_raw; stream-serialized so safe)
    gemm_wmma<2, false><<<dim3(256 / BN, BATCH / BM), blk, 0, stream>>>(mpb, mett, nullptr, Pf, 256, 256);

    // per-row geometry
    geom_rows<<<BATCH / 8, blk, 0, stream>>>(out + OUT_MP, Pf, center, cm, gfb, spart);

    // decoder (g1 reuses h region)
    gemm_wmma<0, false><<<dim3(512 / BN, BATCH / BM), blk, 0, stream>>>(gfb, w3t, b3, h,   512, 256);
    gemm_wmma<1, false><<<dim3(256 / BN, BATCH / BM), blk, 0, stream>>>(h,   w4t, b4, out, 256, 512);

    // scalars
    finalize_stats<<<1, 64, 0, stream>>>(spart, connp, out, BATCH / 8, 1024);
}
